// SSMOperationBaseline_12661563589164
// MI455X (gfx1250) — compile-verified, matrix operand order checked by reading, not run on hardware
//
#include <hip/hip_runtime.h>
#include <hip/hip_bf16.h>

typedef __attribute__((ext_vector_type(16))) _Float16 v16h;
typedef __attribute__((ext_vector_type(8)))  float    v8f;
typedef __attribute__((ext_vector_type(4)))  float    v4f;

#define B_    1024
#define T_    2048
#define F_    32
#define H_    64
#define NOPS_ 8

#define NCHUNK 16                 // chunks per batch row
#define TC     (T_ / NCHUNK)      // 128 timesteps per chunk
#define NTILE  (TC / 16)          // 8 WMMA tiles of 16 timesteps per chunk

// ---------------------------------------------------------------------------
// Branch-free activations built on CDNA5 hardware transcendentals.
// ---------------------------------------------------------------------------
__device__ __forceinline__ float fast_sigmoid(float x) {
    // v_exp_f32 + v_add + v_rcp_f32 : branch-free, TRANS32 co-executes w/ VALU
    return __builtin_amdgcn_rcpf(1.0f + __expf(-x));
}

__device__ __forceinline__ float fast_tanh(float x) {
#if __has_builtin(__builtin_amdgcn_tanhf)
    return __builtin_amdgcn_tanhf(x);      // V_TANH_F32 (CDNA5 TRANS32 op)
#else
    // tanh(x) = 1 - 2/(exp(2x)+1), branch-free
    return 1.0f - 2.0f * __builtin_amdgcn_rcpf(1.0f + __expf(2.0f * x));
#endif
}

// ---------------------------------------------------------------------------
// Zero the hT accumulator workspace (harness poisons d_ws with 0xAA).
// ---------------------------------------------------------------------------
__global__ void ssm_zero_ws(float* __restrict__ p, int n) {
    int i = blockIdx.x * blockDim.x + threadIdx.x;
    if (i < n) p[i] = 0.0f;
}

// ---------------------------------------------------------------------------
// Main kernel: one wave32 per (chunk, batch). Computes
//   S[b,h] += d^(T - t_end) * sum_{t in chunk} d^(t_end-1-t) * g_t * u_t
// using v_wmma_f32_16x16x32_f16 for the [16x32]x[32x16] projections.
// The time-invariant decay turns the scan into a weighted sum, so chunks
// are fully parallel and combine with a single scale + atomicAdd.
// ---------------------------------------------------------------------------
__global__ __launch_bounds__(32) void ssm_chunk_kernel(
    const float* __restrict__ events,   // [B, T, F]
    const float* __restrict__ W_in,     // [F, H]
    const float* __restrict__ b_in,     // [H]
    const float* __restrict__ decay,    // [H]
    const float* __restrict__ W_gate,   // [F, H]
    const float* __restrict__ b_gate,   // [H]
    float* __restrict__ hT)             // [B, H] accumulator (workspace)
{
    const int lane  = threadIdx.x & 31;
    const int chunk = blockIdx.x;       // 0..NCHUNK-1
    const int b     = blockIdx.y;       // 0..B-1
    const int t0    = chunk * TC;

    const int m     = lane & 15;        // A-row (timestep-in-tile) / D-column (h-in-tile)
    const int khalf = lane >> 4;        // which K half this lane holds
    const int kb    = khalf * 8;        // K base: {0..7,16..23} or {8..15,24..31}

    // ---- Load B operands (W_in / W_gate) once per wave, per ISA 16-bit
    //      B-matrix layout: lane holds column n = m, element i holds
    //      K = kb + (i&7) + ((i>>3)<<4).  Weights are tiny -> L2 resident.
    v16h Bin[4], Bga[4];
    float dd[4], d8a[4], d16a[4], binh[4], bgah[4];
    #pragma unroll
    for (int nt = 0; nt < 4; ++nt) {
        const int h = nt * 16 + m;
        v16h bi, bg;
        #pragma unroll
        for (int i = 0; i < 16; ++i) {
            const int K = kb + (i & 7) + ((i >> 3) << 4);
            bi[i] = (_Float16)W_in [K * H_ + h];
            bg[i] = (_Float16)W_gate[K * H_ + h];
        }
        Bin[nt] = bi;
        Bga[nt] = bg;
        binh[nt] = b_in[h];
        bgah[nt] = b_gate[h];
        const float d  = fast_sigmoid(decay[h]);
        const float d2 = d * d, d4 = d2 * d2, d8 = d4 * d4;
        dd[nt] = d; d8a[nt] = d8; d16a[nt] = d8 * d8;
    }

    // ---- Stream over tiles of 16 timesteps ----
    float S[4] = {0.f, 0.f, 0.f, 0.f};

    for (int tile = 0; tile < NTILE; ++tile) {
        const int trow = t0 + tile * 16 + m;      // this lane's timestep row
        const float* row = events + ((size_t)b * T_ + trow) * F_;

        // prefetch next tile's row (global_prefetch_b8)
        if (tile + 1 < NTILE) __builtin_prefetch(row + 16 * F_, 0, 1);

        // A operand: 16-bit A-matrix 16x32 layout. Lane holds row m,
        // K = {kb..kb+7, kb+16..kb+23}: two 32B runs -> four b128 loads.
        // events is a 256MB single-pass stream (> 192MB L2): non-temporal.
        const v4f r0 = __builtin_nontemporal_load((const v4f*)(row + kb));
        const v4f r1 = __builtin_nontemporal_load((const v4f*)(row + kb + 4));
        const v4f r2 = __builtin_nontemporal_load((const v4f*)(row + kb + 16));
        const v4f r3 = __builtin_nontemporal_load((const v4f*)(row + kb + 20));
        v16h A;
        #pragma unroll
        for (int i = 0; i < 4; ++i) {
            A[i]      = (_Float16)r0[i];
            A[4 + i]  = (_Float16)r1[i];
            A[8 + i]  = (_Float16)r2[i];
            A[12 + i] = (_Float16)r3[i];
        }

        #pragma unroll
        for (int nt = 0; nt < 4; ++nt) {
            // Accumulators seeded with the bias (same bias for every row).
            v8f cu, cg;
            #pragma unroll
            for (int r = 0; r < 8; ++r) { cu[r] = binh[nt]; cg[r] = bgah[nt]; }

            cu = __builtin_amdgcn_wmma_f32_16x16x32_f16(
                    false, A, false, Bin[nt], (short)0, cu, false, false);
            cg = __builtin_amdgcn_wmma_f32_16x16x32_f16(
                    false, A, false, Bga[nt], (short)0, cg, false, false);

            // Weighted row reduction: VGPR r holds row (m = r + 8*khalf);
            // weight d^(15 - m) = base * d^(7-r), base = (khalf==0) ? d^8 : 1.
            const float d = dd[nt];
            float w = khalf ? 1.0f : d8a[nt];   // weight at r = 7
            float P = 0.0f;
            #pragma unroll
            for (int r = 7; r >= 0; --r) {
                const float u = fast_tanh(cu[r]);
                const float g = fast_sigmoid(cg[r]);
                P = fmaf(w, g * u, P);
                w *= d;
            }
            // local recurrence across tiles: S = d^16 * S + P
            S[nt] = fmaf(d16a[nt], S[nt], P);
        }
    }

    // ---- Combine: scale local sum by d^(T - t_end), accumulate globally.
    // Lane L and L+16 hold the two row-half partials for the same h; both
    // atomically add to the same address.
    const float rem = (float)(T_ - (t0 + TC));
    #pragma unroll
    for (int nt = 0; nt < 4; ++nt) {
        const int h = nt * 16 + m;
        const float sc = __powf(dd[nt], rem);   // d in (0,1]; pow(d,0)=1
        atomicAdd(&hT[(size_t)b * H_ + h], S[nt] * sc);
    }
}

// ---------------------------------------------------------------------------
// Head: out[b, o] = hT[b, :] . W_head[:, o] + b_head[o]   (1024 x 8 outputs)
// ---------------------------------------------------------------------------
__global__ void ssm_head_kernel(const float* __restrict__ hT,
                                const float* __restrict__ W_head,  // [H, NOPS]
                                const float* __restrict__ b_head,  // [NOPS]
                                float* __restrict__ out)           // [B, NOPS]
{
    const int idx = blockIdx.x * blockDim.x + threadIdx.x;
    if (idx >= B_ * NOPS_) return;
    const int b = idx / NOPS_;
    const int o = idx % NOPS_;
    float acc = b_head[o];
    #pragma unroll
    for (int h = 0; h < H_; ++h)
        acc = fmaf(hT[(size_t)b * H_ + h], W_head[h * NOPS_ + o], acc);
    out[idx] = acc;
}

// ---------------------------------------------------------------------------
extern "C" void kernel_launch(void* const* d_in, const int* in_sizes, int n_in,
                              void* d_out, int out_size, void* d_ws, size_t ws_size,
                              hipStream_t stream) {
    const float* events = (const float*)d_in[0];
    const float* W_in   = (const float*)d_in[1];
    const float* b_in   = (const float*)d_in[2];
    const float* decay  = (const float*)d_in[3];
    const float* W_gate = (const float*)d_in[4];
    const float* b_gate = (const float*)d_in[5];
    const float* W_head = (const float*)d_in[6];
    const float* b_head = (const float*)d_in[7];
    float* out = (float*)d_out;
    float* hT  = (float*)d_ws;          // B*H floats = 256 KB accumulator

    // 1) zero the accumulator workspace
    {
        const int n = B_ * H_;
        ssm_zero_ws<<<(n + 255) / 256, 256, 0, stream>>>(hT, n);
    }
    // 2) chunked WMMA projections + weighted recurrence reduction
    {
        dim3 grid(NCHUNK, B_);
        ssm_chunk_kernel<<<grid, 32, 0, stream>>>(events, W_in, b_in, decay,
                                                  W_gate, b_gate, hT);
    }
    // 3) tiny head GEMM
    {
        const int n = B_ * NOPS_;
        ssm_head_kernel<<<(n + 255) / 256, 256, 0, stream>>>(hT, W_head, b_head, out);
    }
}
